// LzScaleDotAttention_38190849196228
// MI455X (gfx1250) — matrix operand order repaired and checked by reading, not compiled, not run on hardware
//
#include <hip/hip_runtime.h>

// ---- types ----
typedef _Float16 h8   __attribute__((ext_vector_type(8)));
typedef _Float16 h16  __attribute__((ext_vector_type(16)));
typedef float    f8   __attribute__((ext_vector_type(8)));
typedef unsigned int u32x4 __attribute__((ext_vector_type(4)));
typedef int          i32x8 __attribute__((ext_vector_type(8)));
typedef int          i32x4 __attribute__((ext_vector_type(4)));

union H16u { h16 v; h8 h[2]; };

#define BATCH 8
#define SEQ   2048
#define DIM   512
#define BM    64            // output rows (k-index) per block
#define BJ    32            // contraction chunk (q-index)
#define NITER (SEQ / BJ)
#define NTOT  (BATCH * SEQ * DIM)

// eps * sqrt(D): reference adds eps AFTER the 1/sqrt(d) scaling; we keep P
// unscaled (pure exp) so the scale folds into eps: 1e-7 * sqrt(512).
#define EPS_SCALED 2.2627417e-6f

#if __has_builtin(__builtin_amdgcn_tensor_load_to_lds) && __has_builtin(__builtin_amdgcn_s_wait_tensorcnt)
#define HAVE_TDM 1
#else
#define HAVE_TDM 0
#endif

// ---------------- pre-pass kernels (f16 path) ----------------

// elementwise f32 -> f16 (row-major, for q and k)
__global__ __launch_bounds__(256)
void cvt_f16_kernel(const float* __restrict__ src, _Float16* __restrict__ dst) {
    const int i = (blockIdx.x * 256 + threadIdx.x) * 8;
    float4 a = *(const float4*)(src + i);
    float4 b = *(const float4*)(src + i + 4);
    h8 o;
    o[0] = (_Float16)a.x; o[1] = (_Float16)a.y; o[2] = (_Float16)a.z; o[3] = (_Float16)a.w;
    o[4] = (_Float16)b.x; o[5] = (_Float16)b.y; o[6] = (_Float16)b.z; o[7] = (_Float16)b.w;
    *(h8*)(dst + i) = o;
}

// v (B, L, D) f32 -> per-chunk transposed f16: vTg[b][chunk][d][j] = v[b][chunk*32+j][d]
__global__ __launch_bounds__(256)
void cvt_vT_kernel(const float* __restrict__ v, _Float16* __restrict__ vTg) {
    const int t    = blockIdx.x * 256 + threadIdx.x;
    const int dseg = t & 31;           // 32 segments of 16 d's
    const int jg   = (t >> 5) & (SEQ - 1);
    const int b    = t >> 16;          // SEQ*32 = 65536 threads per batch
    const int it   = jg >> 5;
    const int j    = jg & (BJ - 1);
    const float* src = v + (size_t)(b * SEQ + jg) * DIM + dseg * 16;
    _Float16* dst = vTg + ((size_t)(b * NITER + it) * DIM + dseg * 16) * BJ + j;
    #pragma unroll
    for (int u = 0; u < 4; ++u) {
        float4 x = *(const float4*)(src + u * 4);
        dst[(u * 4 + 0) * BJ] = (_Float16)x.x;
        dst[(u * 4 + 1) * BJ] = (_Float16)x.y;
        dst[(u * 4 + 2) * BJ] = (_Float16)x.z;
        dst[(u * 4 + 3) * BJ] = (_Float16)x.w;
    }
}

// ---------------- main attention kernel ----------------

template <bool F16IN>
__global__ __launch_bounds__(256)
void lz_attn_kernel(const float* __restrict__ qg, const float* __restrict__ kg,
                    const float* __restrict__ vg,
                    const _Float16* __restrict__ q16, const _Float16* __restrict__ k16,
                    const _Float16* __restrict__ vTg,
                    float* __restrict__ outg)
{
    // V chunk, transposed: vT[d][j], f16  (32 KB)
    __shared__ __align__(16) _Float16 vT[DIM * BJ];
    // P tiles (exp(S)) per M-group: 16x32 f16 each (4 KB)
    __shared__ __align__(16) _Float16 pbuf[4][16 * BJ];
    __shared__ float    denom_s[BM];
    __shared__ unsigned mask_s[BM];

    const int tid  = threadIdx.x;
    const int lane = tid & 31;
    const int wave = tid >> 5;
    const int mg   = wave & 3;        // M-group: output rows mg*16 .. mg*16+15
    const int dh   = wave >> 2;       // D-half: cols dh*256 .. dh*256+255
    const int b    = blockIdx.y;
    const int i0   = blockIdx.x * BM; // first output row of this block

    const int  l15 = lane & 15;
    const bool hiL = (lane >= 16);

    if (tid < BM) mask_s[tid] = 0u;

    f8 zero8;
    #pragma unroll
    for (int i = 0; i < 8; ++i) zero8[i] = 0.0f;

    f8 acc[16];
    #pragma unroll
    for (int n = 0; n < 16; ++n) acc[n] = zero8;

    f8 dsum = zero8;

    // A-operand rows: the "k" input (plays the query role).
    const float*    kA   = kg  + (size_t)(b * SEQ + i0 + mg * 16 + l15) * DIM + (hiL ? 16 : 0);
    const _Float16* kA16 = k16 + (size_t)(b * SEQ + i0 + mg * 16 + l15) * DIM + (hiL ? 16 : 0);
    // B-operand rows: the "q" input (plays the key role).
    const float*    qB   = qg  + (size_t)(b * SEQ) * DIM + (hiL ? 16 : 0);
    const _Float16* qB16 = q16 + (size_t)(b * SEQ) * DIM + (hiL ? 16 : 0);
    const float* vbase = vg + (size_t)(b * SEQ) * DIM;

    for (int it = 0; it < NITER; ++it) {
        const int j0 = it * BJ;

        __syncthreads();   // previous iteration's vT/pbuf consumers are done

        // ---- stage V chunk (transposed f16) into LDS ----
        if constexpr (F16IN) {
            const _Float16* src = vTg + (size_t)(b * NITER + it) * (DIM * BJ);
#if HAVE_TDM
            // Tensor Data Mover: flat 32KB tile (4096 x 8-byte units) -> LDS
            if (wave == 0) {
                const unsigned long long ga = (unsigned long long)(uintptr_t)src;
                const unsigned lds_off = (unsigned)(uintptr_t)(&vT[0]);
                u32x4 g0 = { 1u,                       // count=1, user descriptor
                             lds_off,                  // lds_addr (bytes)
                             (unsigned)ga,             // global_addr[31:0]
                             (unsigned)(ga >> 32) | 0x80000000u };  // addr[56:32] | type=2
                i32x8 g1 = { (int)0x00030000u,         // data_size=3 (8B), no multicast
                             (int)(4096u << 16),       // tensor_dim0 = 4096 units (lo16)
                             (int)(1u << 16),          // tensor_dim1 = 1
                             (int)(4096u << 16),       // tile_dim0 = 4096
                             1,                        // tile_dim1 = 1
                             4096,                     // tensor_dim0_stride = 4096
                             0, 0 };
                i32x4 g2 = { 0, 0, 0, 0 };
                i32x4 g3 = { 0, 0, 0, 0 };
                i32x8 g4 = { 0, 0, 0, 0, 0, 0, 0, 0 };
                __builtin_amdgcn_tensor_load_to_lds(g0, g1, g2, g3, g4, 0);
                __builtin_amdgcn_s_wait_tensorcnt((short)0);
            }
#else
            // fallback: cooperative contiguous b128 copy
            #pragma unroll
            for (int u = 0; u < 8; ++u) {
                const int idx = (u * 256 + tid) * 8;
                *(h8*)&vT[idx] = *(const h8*)(src + idx);
            }
#endif
        } else {
            // f32 path: load row-major f32, convert + transpose into LDS
            const int jj = tid >> 3;          // 0..31 : local q-row
            const int c0 = (tid & 7) * 64;    // 64 columns per thread
            const float* src = vbase + (size_t)(j0 + jj) * DIM + c0;
            if (it + 1 < NITER)
                __builtin_prefetch(src + (size_t)BJ * DIM, 0, 1); // global_prefetch_b8
            #pragma unroll
            for (int u = 0; u < 16; ++u) {
                float4 x = *(const float4*)(src + u * 4);
                const int d = c0 + u * 4;
                vT[(d + 0) * BJ + jj] = (_Float16)x.x;
                vT[(d + 1) * BJ + jj] = (_Float16)x.y;
                vT[(d + 2) * BJ + jj] = (_Float16)x.z;
                vT[(d + 3) * BJ + jj] = (_Float16)x.w;
            }
        }
        __syncthreads();

        // ---- S = Ktile * Qchunk^T  (dh==0 waves), exp, rowsum, publish P ----
        if (dh == 0) {
            f8 s0 = zero8, s1 = zero8;
            #pragma unroll
            for (int ks = 0; ks < DIM / 32; ++ks) {
                const int col = ks * 32;
                H16u a;
                if constexpr (F16IN) {
                    const _Float16* p = kA16 + col;
                    a.h[0] = *(const h8*)(p);
                    a.h[1] = *(const h8*)(p + 8);
                } else {
                    float4 x0 = *(const float4*)(kA + col + 0);
                    float4 x1 = *(const float4*)(kA + col + 4);
                    float4 x2 = *(const float4*)(kA + col + 8);
                    float4 x3 = *(const float4*)(kA + col + 12);
                    a.v[0]  = (_Float16)x0.x; a.v[1]  = (_Float16)x0.y;
                    a.v[2]  = (_Float16)x0.z; a.v[3]  = (_Float16)x0.w;
                    a.v[4]  = (_Float16)x1.x; a.v[5]  = (_Float16)x1.y;
                    a.v[6]  = (_Float16)x1.z; a.v[7]  = (_Float16)x1.w;
                    a.v[8]  = (_Float16)x2.x; a.v[9]  = (_Float16)x2.y;
                    a.v[10] = (_Float16)x2.z; a.v[11] = (_Float16)x2.w;
                    a.v[12] = (_Float16)x3.x; a.v[13] = (_Float16)x3.y;
                    a.v[14] = (_Float16)x3.z; a.v[15] = (_Float16)x3.w;
                }
                #pragma unroll
                for (int nt = 0; nt < 2; ++nt) {
                    H16u bq;
                    if constexpr (F16IN) {
                        const _Float16* qp = qB16 + (size_t)(j0 + nt * 16 + l15) * DIM + col;
                        bq.h[0] = *(const h8*)(qp);
                        bq.h[1] = *(const h8*)(qp + 8);
                    } else {
                        const float* qp = qB + (size_t)(j0 + nt * 16 + l15) * DIM + col;
                        float4 y0 = *(const float4*)(qp + 0);
                        float4 y1 = *(const float4*)(qp + 4);
                        float4 y2 = *(const float4*)(qp + 8);
                        float4 y3 = *(const float4*)(qp + 12);
                        bq.v[0]  = (_Float16)y0.x; bq.v[1]  = (_Float16)y0.y;
                        bq.v[2]  = (_Float16)y0.z; bq.v[3]  = (_Float16)y0.w;
                        bq.v[4]  = (_Float16)y1.x; bq.v[5]  = (_Float16)y1.y;
                        bq.v[6]  = (_Float16)y1.z; bq.v[7]  = (_Float16)y1.w;
                        bq.v[8]  = (_Float16)y2.x; bq.v[9]  = (_Float16)y2.y;
                        bq.v[10] = (_Float16)y2.z; bq.v[11] = (_Float16)y2.w;
                        bq.v[12] = (_Float16)y3.x; bq.v[13] = (_Float16)y3.y;
                        bq.v[14] = (_Float16)y3.z; bq.v[15] = (_Float16)y3.w;
                    }
                    if (nt == 0)
                        s0 = __builtin_amdgcn_wmma_f32_16x16x32_f16(
                                 false, a.v, false, bq.v, (short)0, s0, false, false);
                    else
                        s1 = __builtin_amdgcn_wmma_f32_16x16x32_f16(
                                 false, a.v, false, bq.v, (short)0, s1, false, false);
                }
            }
            f8 e0, e1;
            #pragma unroll
            for (int i = 0; i < 8; ++i) { e0[i] = __expf(s0[i]); e1[i] = __expf(s1[i]); }

            _Float16* pb = pbuf[mg];
            #pragma unroll
            for (int i = 0; i < 8; ++i) {
                const int row = i + (hiL ? 8 : 0);
                pb[row * BJ + l15]      = (_Float16)e0[i];
                pb[row * BJ + 16 + l15] = (_Float16)e1[i];
            }
            #pragma unroll
            for (int i = 0; i < 8; ++i) {
                float t = e0[i] + e1[i];
                t += __shfl_xor(t, 1);
                t += __shfl_xor(t, 2);
                t += __shfl_xor(t, 4);
                t += __shfl_xor(t, 8);
                dsum[i] += t;
            }
        }
        __syncthreads();

        // ---- PV: acc += P * Vchunk  (all 8 waves) ----
        {
            const _Float16* pr = pbuf[mg] + l15 * BJ + (hiL ? 16 : 0);
            H16u ap;
            ap.h[0] = *(const h8*)(pr);
            ap.h[1] = *(const h8*)(pr + 8);
            #pragma unroll
            for (int nt = 0; nt < 16; ++nt) {
                const int d = dh * 256 + nt * 16 + l15;
                const _Float16* vp = &vT[d * BJ + (hiL ? 16 : 0)];
                H16u bv;
                bv.h[0] = *(const h8*)(vp);
                bv.h[1] = *(const h8*)(vp + 8);
                acc[nt] = __builtin_amdgcn_wmma_f32_16x16x32_f16(
                              false, ap.v, false, bv.v, (short)0, acc[nt], false, false);
            }
        }
    }

    // ---- denominators to LDS ----
    if (dh == 0 && (lane == 0 || lane == 16)) {
        const int rbase = mg * 16 + (hiL ? 8 : 0);
        #pragma unroll
        for (int i = 0; i < 8; ++i) denom_s[rbase + i] = dsum[i];
    }

    // ---- mask pass: any(v[b, i0+row, :] != 0) ----
    {
        const int row = tid >> 2;
        const int seg = tid & 3;
        const float* src = vg + (size_t)(b * SEQ + i0 + row) * DIM + seg * 128;
        unsigned nz = 0u;
        #pragma unroll 8
        for (int u = 0; u < 32; ++u) {
            float4 x = *(const float4*)(src + u * 4);
            nz |= (unsigned)((x.x != 0.f) | (x.y != 0.f) | (x.z != 0.f) | (x.w != 0.f));
        }
        if (nz) atomicOr(&mask_s[row], 1u);
    }
    __syncthreads();

    // ---- scale + store ----
    float sc[8];
    #pragma unroll
    for (int i = 0; i < 8; ++i) {
        const int row = mg * 16 + i + (hiL ? 8 : 0);
        sc[i] = mask_s[row] ? (1.0f / (denom_s[row] + EPS_SCALED)) : 0.0f;
    }
    float* orow = outg + (size_t)(b * SEQ + i0 + mg * 16 + (hiL ? 8 : 0)) * DIM
                       + dh * 256 + l15;
    #pragma unroll
    for (int nt = 0; nt < 16; ++nt) {
        #pragma unroll
        for (int i = 0; i < 8; ++i) {
            orow[(size_t)i * DIM + nt * 16] = acc[nt][i] * sc[i];
        }
    }
}

extern "C" void kernel_launch(void* const* d_in, const int* in_sizes, int n_in,
                              void* d_out, int out_size, void* d_ws, size_t ws_size,
                              hipStream_t stream) {
    (void)in_sizes; (void)n_in; (void)out_size;
    const float* q = (const float*)d_in[0];
    const float* k = (const float*)d_in[1];
    const float* v = (const float*)d_in[2];
    float* out = (float*)d_out;

    _Float16* q16  = (_Float16*)d_ws;
    _Float16* k16  = q16 + (size_t)NTOT;
    _Float16* vT16 = k16 + (size_t)NTOT;
    const size_t need = (size_t)3 * NTOT * sizeof(_Float16);

    dim3 grid(SEQ / BM, BATCH);
    if (d_ws != nullptr && ws_size >= need) {
        cvt_f16_kernel<<<dim3(NTOT / (256 * 8)), dim3(256), 0, stream>>>(q, q16);
        cvt_f16_kernel<<<dim3(NTOT / (256 * 8)), dim3(256), 0, stream>>>(k, k16);
        cvt_vT_kernel<<<dim3(BATCH * SEQ * 32 / 256), dim3(256), 0, stream>>>(v, vT16);
        lz_attn_kernel<true><<<grid, dim3(256), 0, stream>>>(q, k, v, q16, k16, vT16, out);
    } else {
        lz_attn_kernel<false><<<grid, dim3(256), 0, stream>>>(q, k, v, q16, k16, vT16, out);
    }
}
